// HardAndLayer_12884901888483
// MI455X (gfx1250) — matrix operand order repaired and checked by reading, not compiled, not run on hardware
//
#include <hip/hip_runtime.h>

// CDNA5 / gfx1250: wave32, WMMA 16x16x32 f16 -> f32 accumulate.
typedef _Float16 v16h __attribute__((ext_vector_type(16)));
typedef float    v8f  __attribute__((ext_vector_type(8)));

// out[o] = (sum_i w[o,i] * (x[i] ? 0 : 1)) == 0
// Weights are exactly {0.0f, 1.0f}; counts <= 8192 -> exact in f16 inputs / f32 accum.
//
// Block: 256 threads = 8 wave32s, one 16-row output tile per block.
// Each wave handles a contiguous K-chunk of IN/8, unrolled 2x (K=64/iter),
// and accumulates a 16x16 WMMA tile whose 16 columns are identical
// (B columns duplicated), so extraction only needs the documented C/D row
// layout: VGPR r = row r (lanes 0-15) / row r+8 (lanes 16-31).
__global__ __launch_bounds__(256, 4)
void hard_and_wmma_kernel(const float* __restrict__ W,
                          const unsigned char* __restrict__ X,
                          float* __restrict__ out,
                          int OUT, int IN)
{
    __shared__ _Float16 nxh[8192];   // negated input as f16 (16 KB of 320 KB LDS)
    __shared__ float    partial[16]; // cross-wave row sums

    const int t    = threadIdx.x;
    const int lane = t & 31;
    // Force the wave id scalar so the K-loop bounds live in SGPRs and the
    // backend emits a plain scalar-controlled loop (no exec-masked vector loop).
    const int wave = __builtin_amdgcn_readfirstlane(t >> 5);   // 0..7, uniform
    const int row0 = blockIdx.x * 16;

    // nx[i] = x[i] ? 0 : 1 (exact in f16). x is numpy bool -> 1 byte each.
    for (int i = t; i < IN; i += 256)
        nxh[i] = X[i] ? (_Float16)0.0f : (_Float16)1.0f;
    if (t < 16) partial[t] = 0.0f;
    __syncthreads();

    // ISA 16-bit A 16x32 layout: lanes 0-15 hold M=0..15, K = h + (h<8?0:8);
    // lanes 16-31 hold the same M with K shifted by +8.
    // This code's A and B (KxN) order: lanes 0-15 hold K=0..15, lanes 16-31 K=16..31.
    const int m  = lane & 15;
    const int hi = lane >> 4;
    const int c0 = hi * 16;                // A: float offset inside K=32 tile
    const int b0 = hi * 16;                // B: half  offset inside K=32 tile

    const int kChunk = IN >> 3;            // per-wave K range (1024 for IN=8192)
    const int kBeg   = wave * kChunk;      // scalar
    const int kEnd   = kBeg + kChunk;      // scalar

    const float* rowPtr = W + (size_t)(row0 + m) * (size_t)IN;

    v8f acc = {};
    for (int kb = kBeg; kb < kEnd; kb += 64) {   // 2 WMMAs per trip
        const float* p = rowPtr + kb + c0;

        // Sub-tile 0: floats [c0, c0+16) of this row.
        float4 f0 = *(const float4*)(p + 0);
        float4 f1 = *(const float4*)(p + 4);
        float4 f2 = *(const float4*)(p + 8);
        float4 f3 = *(const float4*)(p + 12);
        // Sub-tile 1: same pattern, 32 floats further on.
        float4 g0 = *(const float4*)(p + 32);
        float4 g1 = *(const float4*)(p + 36);
        float4 g2 = *(const float4*)(p + 40);
        float4 g3 = *(const float4*)(p + 44);

        // Unconditional speculative stream-ahead prefetch (~2 KB ahead).
        // Translation failures past the end of the buffer are silently dropped.
        __builtin_prefetch(p + 512, 0, 0);

        v16h a0, a1;
        a0[0]  = (_Float16)f0.x; a0[1]  = (_Float16)f0.y;
        a0[2]  = (_Float16)f0.z; a0[3]  = (_Float16)f0.w;
        a0[4]  = (_Float16)f1.x; a0[5]  = (_Float16)f1.y;
        a0[6]  = (_Float16)f1.z; a0[7]  = (_Float16)f1.w;
        a0[8]  = (_Float16)f2.x; a0[9]  = (_Float16)f2.y;
        a0[10] = (_Float16)f2.z; a0[11] = (_Float16)f2.w;
        a0[12] = (_Float16)f3.x; a0[13] = (_Float16)f3.y;
        a0[14] = (_Float16)f3.z; a0[15] = (_Float16)f3.w;

        a1[0]  = (_Float16)g0.x; a1[1]  = (_Float16)g0.y;
        a1[2]  = (_Float16)g0.z; a1[3]  = (_Float16)g0.w;
        a1[4]  = (_Float16)g1.x; a1[5]  = (_Float16)g1.y;
        a1[6]  = (_Float16)g1.z; a1[7]  = (_Float16)g1.w;
        a1[8]  = (_Float16)g2.x; a1[9]  = (_Float16)g2.y;
        a1[10] = (_Float16)g2.z; a1[11] = (_Float16)g2.w;
        a1[12] = (_Float16)g3.x; a1[13] = (_Float16)g3.y;
        a1[14] = (_Float16)g3.z; a1[15] = (_Float16)g3.w;

        // B fragments from LDS: 16 consecutive halves each, 32B aligned
        // -> 2x ds_load_b128 per fragment. Value depends only on K, so all
        // 16 columns of B are identical.
        v16h b0v = *(const v16h*)(&nxh[kb + b0]);
        v16h b1v = *(const v16h*)(&nxh[kb + 32 + b0]);

        // D = A x B + C   (v_wmma_f32_16x16x32_f16)
        acc = __builtin_amdgcn_wmma_f32_16x16x32_f16(
                false, a0, false, b0v, (short)0, acc, false, false);
        acc = __builtin_amdgcn_wmma_f32_16x16x32_f16(
                false, a1, false, b1v, (short)0, acc, false, false);
    }

    // All columns identical: lane 0 carries rows 0..7, lane 16 carries rows 8..15.
    if (lane == 0) {
        #pragma unroll
        for (int r = 0; r < 8; ++r) atomicAdd(&partial[r], acc[r]);
    } else if (lane == 16) {
        #pragma unroll
        for (int r = 0; r < 8; ++r) atomicAdd(&partial[8 + r], acc[r]);
    }
    __syncthreads();

    if (t < 16 && (row0 + t) < OUT)
        out[row0 + t] = (partial[t] == 0.0f) ? 1.0f : 0.0f;
}

extern "C" void kernel_launch(void* const* d_in, const int* in_sizes, int n_in,
                              void* d_out, int out_size, void* d_ws, size_t ws_size,
                              hipStream_t stream) {
    (void)n_in; (void)d_ws; (void)ws_size;
    const float*         W = (const float*)d_in[0];          // [OUT, IN] fp32 {0,1}
    const unsigned char* X = (const unsigned char*)d_in[1];  // [IN] bool (1 byte)
    float* out = (float*)d_out;                              // [OUT] bool -> float

    const int IN  = in_sizes[1];
    const int OUT = in_sizes[0] / IN;

    const int blocks = (OUT + 15) / 16;   // 512 blocks for OUT=8192
    hipLaunchKernelGGL(hard_and_wmma_kernel, dim3(blocks), dim3(256), 0, stream,
                       W, X, out, OUT, IN);
    (void)out_size;
}
